// SelfAttention_42820823941612
// MI455X (gfx1250) — matrix operand order repaired and checked by reading, not compiled, not run on hardware
//
#include <hip/hip_runtime.h>

// ---------------------------------------------------------------------------
// Self-attention for MI455X (gfx1250): bf16 WMMA GEMMs with async-to-LDS
// double-buffered staging; f32 softmax.
//   x:[4,2048,1024] f32, Wq/Wk/Wv:[1024,1024] f32
//   Q=x@Wq^T, K=x@Wk^T, V^T stored at projection, S=Q@K^T, P=softmax(S), O=P@V
// ---------------------------------------------------------------------------

typedef __attribute__((ext_vector_type(16))) __bf16 v16bf;
typedef __attribute__((ext_vector_type(8)))  float  v8f;

#define BATCH 4
#define SEQ   2048
#define HID   1024

#define BK    64            // K-elements staged per tile (2 WMMA k-steps)
#define LPK   72            // LDS row pitch (bf16 elems): 144B, 36 banks, 16B aligned
#define MATB  (128 * LPK * 2)       // bytes per matrix per buffer
#define BUFB  (2 * MATB)            // bytes per buffer (A + B)

__device__ __forceinline__ unsigned short f2bf(float f) {
    unsigned int u = __float_as_uint(f);
    unsigned int r = (u + 0x7FFFu + ((u >> 16) & 1u)) >> 16;   // round-nearest-even
    return (unsigned short)r;
}

__device__ __forceinline__ v8f wmma_bf16(v16bf a, v16bf b, v8f c) {
    return __builtin_amdgcn_wmma_f32_16x16x32_bf16(false, a, false, b,
                                                   (short)0, c, false, false);
}

union Frag {
    uint4  q[2];
    v16bf  v;
};

// CDNA5 async DMA: global -> LDS, 16B per lane, tracked with ASYNCcnt.
__device__ __forceinline__ void async_b128(unsigned ldsoff, unsigned goff,
                                           const unsigned short* sbase) {
    asm volatile("global_load_async_to_lds_b128 %0, %1, %2"
                 :: "v"(ldsoff), "v"(goff), "s"(sbase) : "memory");
}
__device__ __forceinline__ void wait_async0() {
    asm volatile("s_wait_asynccnt 0x0" ::: "memory");
}

// ---------------------------------------------------------------------------
// f32 -> bf16 conversion
// ---------------------------------------------------------------------------
__global__ __launch_bounds__(256)
void convert_f32_bf16(const float* __restrict__ in, unsigned short* __restrict__ out, int n) {
    int i = blockIdx.x * 256 + threadIdx.x;
    if (i < n) out[i] = f2bf(in[i]);
}

// ---------------------------------------------------------------------------
// Tiled WMMA GEMM:  C[M,N] = A[M,K] * B^T,  A:[M,K] row-major, B:[N,K] row-major.
// Block = 256 threads (8 waves), tile 128x128, staged K-step 64, double-buffered
// async-to-LDS. Wave grid 2(M) x 4(N); each wave: 4x2 tiles -> 16 WMMAs / tile.
//   OUT_BF16 : write bf16 C (Q/K projections)
//   OUT_TRANS: write bf16 C transposed per batch as [batch][N][SEQ] (V^T)
// blockIdx.z selects batch/weight via strides.
// ---------------------------------------------------------------------------
template<bool OUT_BF16, bool OUT_TRANS>
__global__ __launch_bounds__(256)
void gemm_wmma(const unsigned short* __restrict__ A,
               const unsigned short* __restrict__ B,
               float* __restrict__ Cf,
               unsigned short* __restrict__ Cb,
               int M, int N, int Kd,
               long long aStride, long long bStride, long long cStride) {
    __shared__ __align__(16) unsigned short lds[2][2][128][LPK];

    const int tid   = threadIdx.x;
    const int lane  = tid & 31;
    const int wave  = tid >> 5;
    const int waveM = wave >> 2;        // 0..1
    const int waveN = wave & 3;         // 0..3
    const int half  = lane >> 4;        // ISA fragment lane-half
    const int l16   = lane & 15;

    const int mBase = blockIdx.y * 128;
    const int nBase = blockIdx.x * 128;
    const unsigned short* Ab = A + (long long)blockIdx.z * aStride;
    const unsigned short* Bb = B + (long long)blockIdx.z * bStride;

    // Per-thread staging slots: 4 x b128 per matrix per tile (128 rows x 8 uint4).
    // Instruction i covers u = tid + i*256 -> contiguous 4KB per instruction.
    unsigned gA[4], gB[4], lA[4], lB[4];
    const unsigned ldsBase = (unsigned)(unsigned long long)(uintptr_t)(&lds[0][0][0][0]);
#pragma unroll
    for (int i = 0; i < 4; ++i) {
        int u   = tid + i * 256;
        int row = u >> 3;               // 0..127
        int c8  = u & 7;                // uint4 within 64-wide row
        gA[i] = (unsigned)(((mBase + row) * Kd + c8 * 8) * 2);
        gB[i] = (unsigned)(((nBase + row) * Kd + c8 * 8) * 2);
        lA[i] = ldsBase + (unsigned)(row * (LPK * 2) + c8 * 16);
        lB[i] = lA[i] + MATB;
    }

    v8f acc[4][2];
#pragma unroll
    for (int mi = 0; mi < 4; ++mi)
#pragma unroll
        for (int ni = 0; ni < 2; ++ni)
            acc[mi][ni] = (v8f){0.f, 0.f, 0.f, 0.f, 0.f, 0.f, 0.f, 0.f};

    const int ntiles = Kd / BK;

    // Prologue: stage tile 0 into buffer 0.
#pragma unroll
    for (int i = 0; i < 4; ++i) async_b128(lA[i], gA[i], Ab);
#pragma unroll
    for (int i = 0; i < 4; ++i) async_b128(lB[i], gB[i], Bb);

    for (int t = 0; t < ntiles; ++t) {
        wait_async0();          // this wave's tile-t DMA done
        __syncthreads();        // everyone's tile-t DMA done; everyone done reading other buf

        if (t + 1 < ntiles) {   // stage tile t+1 into the other buffer (DMA overlaps compute)
            unsigned kb   = (unsigned)((t + 1) * BK * 2);
            unsigned bofs = (unsigned)(((t + 1) & 1) * BUFB);
#pragma unroll
            for (int i = 0; i < 4; ++i) async_b128(lA[i] + bofs, gA[i] + kb, Ab);
#pragma unroll
            for (int i = 0; i < 4; ++i) async_b128(lB[i] + bofs, gB[i] + kb, Bb);
        }

        const unsigned short (*As)[LPK] = lds[t & 1][0];
        const unsigned short (*Bs)[LPK] = lds[t & 1][1];
#pragma unroll
        for (int sk = 0; sk < 2; ++sk) {
            const int ko = sk * 32;
            v16bf bfrag[2];
#pragma unroll
            for (int ni = 0; ni < 2; ++ni) {
                int row = waveN * 32 + ni * 16 + l16;
                Frag f;
                f.q[0] = *(const uint4*)&Bs[row][ko + half * 16];     // lanes 0-15: K 0..15
                f.q[1] = *(const uint4*)&Bs[row][ko + half * 16 + 8]; // lanes 16-31: K 16..31
                bfrag[ni] = f.v;
            }
#pragma unroll
            for (int mi = 0; mi < 4; ++mi) {
                int row = waveM * 64 + mi * 16 + l16;
                Frag f;
                f.q[0] = *(const uint4*)&As[row][ko + half * 16];     // K 0..7 / 16..23
                f.q[1] = *(const uint4*)&As[row][ko + half * 16 + 8]; // K 8..15 / 24..31
                v16bf a = f.v;
#pragma unroll
                for (int ni = 0; ni < 2; ++ni)
                    acc[mi][ni] = wmma_bf16(a, bfrag[ni], acc[mi][ni]);
            }
        }
    }

    // Store: C-layout VGPR r -> M = r + 8*half, N = l16.
    const long long cb = (long long)blockIdx.z * cStride;
#pragma unroll
    for (int mi = 0; mi < 4; ++mi) {
#pragma unroll
        for (int ni = 0; ni < 2; ++ni) {
            int col = nBase + waveN * 32 + ni * 16 + l16;
#pragma unroll
            for (int r = 0; r < 8; ++r) {
                int   row = mBase + waveM * 64 + mi * 16 + half * 8 + r;
                float v   = acc[mi][ni][r];
                if (OUT_TRANS) {
                    // V^T: [batch][n][k], batch = row/SEQ, k = row%SEQ
                    long long idx = ((long long)(row >> 11) * HID + col) * SEQ + (row & (SEQ - 1));
                    Cb[idx] = f2bf(v);
                } else if (OUT_BF16) {
                    Cb[cb + (long long)row * N + col] = f2bf(v);
                } else {
                    Cf[cb + (long long)row * N + col] = v;
                }
            }
        }
    }
}

// ---------------------------------------------------------------------------
// Row softmax over 2048-wide rows: f32 scores in, bf16 probabilities out.
// ---------------------------------------------------------------------------
__global__ __launch_bounds__(256)
void softmax_rows(const float* __restrict__ S, unsigned short* __restrict__ P) {
    __shared__ float red[256];
    const long long row = blockIdx.x;
    const float* s = S + row * (long long)SEQ;

    float v[8];
    float m = -1e30f;
#pragma unroll
    for (int i = 0; i < 8; ++i) {
        v[i] = s[threadIdx.x + i * 256];
        m = fmaxf(m, v[i]);
    }
    red[threadIdx.x] = m;
    __syncthreads();
    for (int off = 128; off > 0; off >>= 1) {
        if (threadIdx.x < off)
            red[threadIdx.x] = fmaxf(red[threadIdx.x], red[threadIdx.x + off]);
        __syncthreads();
    }
    m = red[0];
    __syncthreads();

    float sum = 0.f;
#pragma unroll
    for (int i = 0; i < 8; ++i) {
        v[i] = __expf(v[i] - m);
        sum += v[i];
    }
    red[threadIdx.x] = sum;
    __syncthreads();
    for (int off = 128; off > 0; off >>= 1) {
        if (threadIdx.x < off)
            red[threadIdx.x] += red[threadIdx.x + off];
        __syncthreads();
    }
    float inv = 1.0f / red[0];
#pragma unroll
    for (int i = 0; i < 8; ++i)
        P[row * (long long)SEQ + threadIdx.x + i * 256] = f2bf(v[i] * inv);
}

// ---------------------------------------------------------------------------
// Launch
// ---------------------------------------------------------------------------
extern "C" void kernel_launch(void* const* d_in, const int* in_sizes, int n_in,
                              void* d_out, int out_size, void* d_ws, size_t ws_size,
                              hipStream_t stream) {
    const float* x  = (const float*)d_in[0];
    const float* Wq = (const float*)d_in[1];
    const float* Wk = (const float*)d_in[2];
    const float* Wv = (const float*)d_in[3];
    float* out = (float*)d_out;

    const long long NX = (long long)BATCH * SEQ * HID;   // 8,388,608
    const long long NW = (long long)HID * HID;           // 1,048,576

    unsigned short* xb  = (unsigned short*)d_ws;
    unsigned short* Wqb = xb  + NX;
    unsigned short* Wkb = Wqb + NW;
    unsigned short* Wvb = Wkb + NW;
    unsigned short* Qb  = Wvb + NW;
    unsigned short* Kb  = Qb  + NX;
    unsigned short* Vt  = Kb  + NX;                      // V^T : [4][1024][2048] bf16
    float*          Sc  = (float*)(Vt + NX);             // [4,2048,2048] f32
    unsigned short* Pb  = (unsigned short*)(Sc + (long long)BATCH * SEQ * SEQ);

    // 1) bf16 conversion
    convert_f32_bf16<<<(unsigned)(NX / 256), 256, 0, stream>>>(x,  xb,  (int)NX);
    convert_f32_bf16<<<(unsigned)(NW / 256), 256, 0, stream>>>(Wq, Wqb, (int)NW);
    convert_f32_bf16<<<(unsigned)(NW / 256), 256, 0, stream>>>(Wk, Wkb, (int)NW);
    convert_f32_bf16<<<(unsigned)(NW / 256), 256, 0, stream>>>(Wv, Wvb, (int)NW);

    // 2a) Q,K projections: z = {Wq,Wk} -> {Qb,Kb}
    dim3 gQK(HID / 128, (BATCH * SEQ) / 128, 2);
    gemm_wmma<true, false><<<gQK, 256, 0, stream>>>(
        xb, Wqb, nullptr, Qb, BATCH * SEQ, HID, HID, 0, NW, NX);

    // 2b) V projection, stored transposed per batch -> Vt
    dim3 gV(HID / 128, (BATCH * SEQ) / 128, 1);
    gemm_wmma<true, true><<<gV, 256, 0, stream>>>(
        xb, Wvb, nullptr, Vt, BATCH * SEQ, HID, HID, 0, 0, 0);

    // 3) Scores: per-batch S = Q @ K^T, f32
    dim3 gS(SEQ / 128, SEQ / 128, BATCH);
    gemm_wmma<false, false><<<gS, 256, 0, stream>>>(
        Qb, Kb, Sc, nullptr, SEQ, SEQ, HID,
        (long long)SEQ * HID, (long long)SEQ * HID, (long long)SEQ * SEQ);

    // 4) Softmax rows -> bf16 P
    softmax_rows<<<BATCH * SEQ, 256, 0, stream>>>(Sc, Pb);

    // 5) Output: per-batch O = P @ (V^T)^T via the same N x K B-path
    dim3 gO(HID / 128, SEQ / 128, BATCH);
    gemm_wmma<false, false><<<gO, 256, 0, stream>>>(
        Pb, Vt, out, nullptr, SEQ, HID, SEQ,
        (long long)SEQ * SEQ, (long long)HID * SEQ, (long long)SEQ * HID);
}
